// Decoder_66889820668660
// MI455X (gfx1250) — matrix operand order repaired and checked by reading, not compiled, not run on hardware
//
#include <hip/hip_runtime.h>
#include <hip/hip_bf16.h>

// ---------------------------------------------------------------------------
// Tacotron-style decoder for MI455X (gfx1250, wave32, WMMA).
//
// Sizes: B=64, T_MEL=2000, E=512, M=80, NFPS=2, P=256, H=1024, T_DEC=1000.
// Strategy:
//  * Pack all weights to f16 row-major [N,K] once (~40MB -> resident in L2).
//  * Prenet + memory pooling computed once, in parallel, with WMMA GEMMs.
//  * Sequential scan: 5 kernels/step (GRU1, att-GEMM, GRU2+res, GRU3+res,
//    proj-GEMM), each a 64xNxK f16 WMMA GEMM with fused epilogue.
//  * GRU kernel keeps 4 f32 accumulators per wave-tile (r, z, i_n, h_n) so the
//    PyTorch gate semantics (n = tanh(i_n + r*h_n)) are exact; sigmoid/tanh
//    blend fused in-register; residual adds fused.
//  * h1/h2/h3 ping-pong buffers avoid intra-step RAW hazards.
// ---------------------------------------------------------------------------

typedef _Float16 h16;
typedef __attribute__((ext_vector_type(16))) _Float16 v16h;
typedef __attribute__((ext_vector_type(8)))  _Float16 v8h;
typedef __attribute__((ext_vector_type(8)))  float    v8f;

#define T_MEL_C 2000
#define B_C     64
#define E_C     512
#define M_C     80
#define NFPS_C  2
#define P_C     256
#define H_C     1024
#define TDEC_C  1000
#define MK_C    96          // M=80 zero-padded to a multiple of 32

// ---- WMMA fragment loaders (layouts per CDNA5 ISA 7.12.2) ------------------
// A 16x32 f16: lanes 0-15 -> rows M=0..15 ; element i: K = khalf*8 + i (i<8),
//              K = 16 + khalf*8 + (i-8) (i>=8), khalf = lane>>4.
__device__ inline v16h load_a_frag(const h16* __restrict__ A, int ldA,
                                   int rowBase, int kb) {
  int lane = threadIdx.x & 31;
  int m = lane & 15, kh = lane >> 4;
  const h16* p = A + (size_t)(rowBase + m) * ldA + kb + kh * 8;
  v8h lo = *(const v8h*)p;
  v8h hi = *(const v8h*)(p + 16);
  v16h a;
#pragma unroll
  for (int i = 0; i < 8; ++i) { a[i] = lo[i]; a[i + 8] = hi[i]; }
  return a;
}

// B 32x16 f16 (B[k,n] = W[n,k], W row-major [N,K]): lane -> column n,
// element i: K = khalf*8 + i (i<8), K = 16 + khalf*8 + (i-8) (i>=8) (matches A).
__device__ inline v16h load_b_frag(const h16* __restrict__ W, int ldW,
                                   int colBase, int kb) {
  int lane = threadIdx.x & 31;
  int n = lane & 15, kg = lane >> 4;
  const h16* p = W + (size_t)(colBase + n) * ldW + kb + kg * 8;
  v8h lo = *(const v8h*)p;
  v8h hi = *(const v8h*)(p + 16);
  v16h b;
#pragma unroll
  for (int i = 0; i < 8; ++i) { b[i] = lo[i]; b[i + 8] = hi[i]; }
  return b;
}

__device__ inline v8f wmma_step(v16h a, v16h b, v8f c) {
  return __builtin_amdgcn_wmma_f32_16x16x32_f16(false, a, false, b,
                                                (short)0, c, false, false);
}

__device__ inline float sigmoidf_(float x) { return 1.f / (1.f + __expf(-x)); }

// ---- Prep kernels ----------------------------------------------------------
__global__ void cvt_f16(const float* __restrict__ s, h16* __restrict__ d, int n) {
  int i = blockIdx.x * blockDim.x + threadIdx.x;
  if (i < n) d[i] = (h16)s[i];
}

__global__ void pad_w1(const float* __restrict__ s, h16* __restrict__ d) {
  int i = blockIdx.x * blockDim.x + threadIdx.x;      // [P,96] from [P,80]
  if (i >= P_C * MK_C) return;
  int n = i / MK_C, k = i % MK_C;
  d[i] = (k < M_C) ? (h16)s[n * M_C + k] : (h16)0.f;
}

// di[t,b,k] = (t==0) ? 0 : decoder_inputs[b, 2t-1, k], zero-padded K 80->96
__global__ void build_di(const float* __restrict__ dec, h16* __restrict__ dip) {
  int i = blockIdx.x * blockDim.x + threadIdx.x;
  if (i >= TDEC_C * B_C * MK_C) return;
  int k = i % MK_C; int bt = i / MK_C; int b = bt % B_C; int t = bt / B_C;
  float v = 0.f;
  if (k < M_C && t > 0)
    v = dec[((size_t)b * T_MEL_C + (2 * t - 1)) * M_C + k];
  dip[i] = (h16)v;
}

// mem[t,b,e] = 0.5*(memorys[b,2t,e] + memorys[b,2t+1,e])
__global__ void pool_mem(const float* __restrict__ mem, h16* __restrict__ dst) {
  int i = blockIdx.x * blockDim.x + threadIdx.x;
  if (i >= TDEC_C * B_C * E_C) return;
  int e = i % E_C; int bt = i / E_C; int b = bt % B_C; int t = bt / B_C;
  size_t base = ((size_t)b * T_MEL_C + 2 * t) * E_C + e;
  dst[i] = (h16)(0.5f * (mem[base] + mem[base + E_C]));
}

__global__ void zero_h16(h16* __restrict__ p, int n) {
  int i = blockIdx.x * blockDim.x + threadIdx.x;
  if (i < n) p[i] = (h16)0.f;
}

// ---- Prenet GEMM: out = relu(A @ W^T + b), rows parallel over T_DEC*B ------
// 8 waves/WG, each wave one 16x16 tile: WG covers 128 rows x 16 cols.
__global__ void gemm_relu_f16(const h16* __restrict__ A, int lda, int K,
                              const h16* __restrict__ W,
                              const float* __restrict__ bias,
                              h16* __restrict__ out, int ldo) {
  int wave = threadIdx.x >> 5;
  int lane = threadIdx.x & 31;
  int rowBase = blockIdx.y * 128 + wave * 16;
  int colBase = blockIdx.x * 16;
  v8f acc = {};
  for (int kb = 0; kb < K; kb += 32)
    acc = wmma_step(load_a_frag(A, lda, rowBase, kb),
                    load_b_frag(W, K, colBase, kb), acc);
  int n = colBase + (lane & 15);
  float bv = bias[n];
  int rhi = (lane >> 4) * 8;
#pragma unroll
  for (int v = 0; v < 8; ++v) {
    float x = acc[v] + bv;
    x = x > 0.f ? x : 0.f;
    out[(size_t)(rowBase + rhi + v) * ldo + n] = (h16)x;
  }
}

// ---- Generic 2-segment GEMM (attention / output projection) ----------------
// out[b,n] = [A0|A1][b,:] @ W[n,:] + bias[n] ; f16 or f32 output.
// Grid: N/32 WGs; 8 waves = 4 row-tiles (64 batch rows) x 2 col-tiles.
__global__ void gemm2seg(const h16* __restrict__ A0, int K0, int ld0,
                         const h16* __restrict__ A1, int K1, int ld1,
                         const h16* __restrict__ W, int K,
                         const float* __restrict__ bias,
                         h16* __restrict__ outh, float* __restrict__ outf,
                         int ldo) {
  int wave = threadIdx.x >> 5;
  int rt = wave & 3, ct = wave >> 2;
  int rowBase = rt * 16;
  int colBase = blockIdx.x * 32 + ct * 16;
  int lane = threadIdx.x & 31;
  v8f acc = {};
  for (int kb = 0; kb < K0; kb += 32)
    acc = wmma_step(load_a_frag(A0, ld0, rowBase, kb),
                    load_b_frag(W, K, colBase, kb), acc);
  for (int kb = 0; kb < K1; kb += 32)
    acc = wmma_step(load_a_frag(A1, ld1, rowBase, kb),
                    load_b_frag(W, K, colBase, K0 + kb), acc);
  int n = colBase + (lane & 15);
  float bv = bias[n];
  int rhi = (lane >> 4) * 8;
#pragma unroll
  for (int v = 0; v < 8; ++v) {
    float x = acc[v] + bv;
    int r = rowBase + rhi + v;
    if (outh) outh[(size_t)r * ldo + n] = (h16)x;
    else      outf[(size_t)r * ldo + n] = x;
  }
}

// ---- Fused GRU cell --------------------------------------------------------
// x = [A0|A1] (Kx = K0+K1 cols of wih). r,z use concatenated-K accumulation;
// n gate keeps i_n and h_n separate (PyTorch semantics). Epilogue fuses
// sigmoid/tanh blend, hprev mix, and optional residual (r = h + resid_in).
__global__ void gru_step(const h16* __restrict__ A0, int K0, int ld0,
                         const h16* __restrict__ A1, int K1, int ld1,
                         const h16* __restrict__ hprev,
                         const h16* __restrict__ wih, int Kx,
                         const h16* __restrict__ whh,
                         const float* __restrict__ bih,
                         const float* __restrict__ bhh,
                         h16* __restrict__ hnew,
                         const h16* __restrict__ resid_in,
                         h16* __restrict__ resid_out) {
  int wave = threadIdx.x >> 5;
  int rt = wave & 3, ct = wave >> 2;
  int rowBase = rt * 16;
  int colBase = blockIdx.x * 32 + ct * 16;
  int lane = threadIdx.x & 31;
  v8f aR = {}, aZ = {}, aIN = {}, aHN = {};
  for (int kb = 0; kb < K0; kb += 32) {
    v16h a = load_a_frag(A0, ld0, rowBase, kb);
    aR  = wmma_step(a, load_b_frag(wih, Kx, colBase,           kb), aR);
    aZ  = wmma_step(a, load_b_frag(wih, Kx, colBase + H_C,     kb), aZ);
    aIN = wmma_step(a, load_b_frag(wih, Kx, colBase + 2 * H_C, kb), aIN);
  }
  for (int kb = 0; kb < K1; kb += 32) {
    v16h a = load_a_frag(A1, ld1, rowBase, kb);
    aR  = wmma_step(a, load_b_frag(wih, Kx, colBase,           K0 + kb), aR);
    aZ  = wmma_step(a, load_b_frag(wih, Kx, colBase + H_C,     K0 + kb), aZ);
    aIN = wmma_step(a, load_b_frag(wih, Kx, colBase + 2 * H_C, K0 + kb), aIN);
  }
  for (int kb = 0; kb < H_C; kb += 32) {
    v16h a = load_a_frag(hprev, H_C, rowBase, kb);
    aR  = wmma_step(a, load_b_frag(whh, H_C, colBase,           kb), aR);
    aZ  = wmma_step(a, load_b_frag(whh, H_C, colBase + H_C,     kb), aZ);
    aHN = wmma_step(a, load_b_frag(whh, H_C, colBase + 2 * H_C, kb), aHN);
  }
  int j = colBase + (lane & 15);
  float bR  = bih[j] + bhh[j];
  float bZ  = bih[H_C + j] + bhh[H_C + j];
  float bIN = bih[2 * H_C + j];
  float bHN = bhh[2 * H_C + j];
  int rhi = (lane >> 4) * 8;
#pragma unroll
  for (int v = 0; v < 8; ++v) {
    int r = rowBase + rhi + v;
    float rg = sigmoidf_(aR[v] + bR);
    float zg = sigmoidf_(aZ[v] + bZ);
    float ng = tanhf(aIN[v] + bIN + rg * (aHN[v] + bHN));
    float hp = (float)hprev[(size_t)r * H_C + j];
    float h  = (1.f - zg) * ng + zg * hp;
    hnew[(size_t)r * H_C + j] = (h16)h;
    if (resid_out)
      resid_out[(size_t)r * H_C + j] =
          (h16)(h + (float)resid_in[(size_t)r * H_C + j]);
  }
}

// ---------------------------------------------------------------------------
extern "C" void kernel_launch(void* const* d_in, const int* in_sizes, int n_in,
                              void* d_out, int out_size, void* d_ws,
                              size_t ws_size, hipStream_t stream) {
  (void)in_sizes; (void)n_in; (void)out_size; (void)ws_size;
  const float* memorys = (const float*)d_in[0];
  const float* dec_in  = (const float*)d_in[1];
  /* d_in[2] memory_lengths: unused by reference decode */
  const float* pre_w1 = (const float*)d_in[3];
  const float* pre_b1 = (const float*)d_in[4];
  const float* pre_w2 = (const float*)d_in[5];
  const float* pre_b2 = (const float*)d_in[6];
  const float* g1_wih = (const float*)d_in[7];
  const float* g1_whh = (const float*)d_in[8];
  const float* g1_bih = (const float*)d_in[9];
  const float* g1_bhh = (const float*)d_in[10];
  const float* att_w  = (const float*)d_in[11];
  const float* att_b  = (const float*)d_in[12];
  const float* g2_wih = (const float*)d_in[13];
  const float* g2_whh = (const float*)d_in[14];
  const float* g2_bih = (const float*)d_in[15];
  const float* g2_bhh = (const float*)d_in[16];
  const float* g3_wih = (const float*)d_in[17];
  const float* g3_whh = (const float*)d_in[18];
  const float* g3_bih = (const float*)d_in[19];
  const float* g3_bhh = (const float*)d_in[20];
  const float* proj_w = (const float*)d_in[21];
  const float* proj_b = (const float*)d_in[22];
  float* out = (float*)d_out;

  // Bump allocator over workspace (256B-aligned f16 blocks)
  char* ws = (char*)d_ws;
  size_t off = 0;
  auto alloc = [&](size_t nhalf) -> h16* {
    h16* p = (h16*)(ws + off);
    off = (off + nhalf * sizeof(h16) + 255) & ~(size_t)255;
    return p;
  };
  h16* w_pw1  = alloc((size_t)P_C * MK_C);
  h16* w_pw2  = alloc((size_t)P_C * P_C);
  h16* w_g1ih = alloc((size_t)3 * H_C * (P_C + E_C));
  h16* w_g1hh = alloc((size_t)3 * H_C * H_C);
  h16* w_att  = alloc((size_t)H_C * (H_C + E_C));
  h16* w_g2ih = alloc((size_t)3 * H_C * H_C);
  h16* w_g2hh = alloc((size_t)3 * H_C * H_C);
  h16* w_g3ih = alloc((size_t)3 * H_C * H_C);
  h16* w_g3hh = alloc((size_t)3 * H_C * H_C);
  h16* w_proj = alloc((size_t)M_C * NFPS_C * (H_C + E_C));
  h16* dip    = alloc((size_t)TDEC_C * B_C * MK_C);
  h16* x1     = alloc((size_t)TDEC_C * B_C * P_C);
  h16* xbuf   = alloc((size_t)TDEC_C * B_C * P_C);
  h16* membuf = alloc((size_t)TDEC_C * B_C * E_C);
  h16* hstate = alloc((size_t)6 * B_C * H_C);   // h1/h2/h3 ping-pong
  h16* d2     = alloc((size_t)B_C * H_C);
  h16* r2     = alloc((size_t)B_C * H_C);
  h16* r3     = alloc((size_t)B_C * H_C);

  const int THR = 256;
  auto nb = [](size_t n) { return (unsigned)((n + 255) / 256); };

  // -- Weight packing (once; ~40MB f16 -> stays hot in 192MB L2) --
  pad_w1<<<nb(P_C * MK_C), THR, 0, stream>>>(pre_w1, w_pw1);
  cvt_f16<<<nb((size_t)P_C * P_C), THR, 0, stream>>>(pre_w2, w_pw2, P_C * P_C);
  cvt_f16<<<nb((size_t)3 * H_C * (P_C + E_C)), THR, 0, stream>>>(g1_wih, w_g1ih, 3 * H_C * (P_C + E_C));
  cvt_f16<<<nb((size_t)3 * H_C * H_C), THR, 0, stream>>>(g1_whh, w_g1hh, 3 * H_C * H_C);
  cvt_f16<<<nb((size_t)H_C * (H_C + E_C)), THR, 0, stream>>>(att_w, w_att, H_C * (H_C + E_C));
  cvt_f16<<<nb((size_t)3 * H_C * H_C), THR, 0, stream>>>(g2_wih, w_g2ih, 3 * H_C * H_C);
  cvt_f16<<<nb((size_t)3 * H_C * H_C), THR, 0, stream>>>(g2_whh, w_g2hh, 3 * H_C * H_C);
  cvt_f16<<<nb((size_t)3 * H_C * H_C), THR, 0, stream>>>(g3_wih, w_g3ih, 3 * H_C * H_C);
  cvt_f16<<<nb((size_t)3 * H_C * H_C), THR, 0, stream>>>(g3_whh, w_g3hh, 3 * H_C * H_C);
  cvt_f16<<<nb((size_t)M_C * NFPS_C * (H_C + E_C)), THR, 0, stream>>>(proj_w, w_proj, M_C * NFPS_C * (H_C + E_C));

  // -- Parallel pre-loop work --
  build_di<<<nb((size_t)TDEC_C * B_C * MK_C), THR, 0, stream>>>(dec_in, dip);
  pool_mem<<<nb((size_t)TDEC_C * B_C * E_C), THR, 0, stream>>>(memorys, membuf);
  zero_h16<<<nb((size_t)6 * B_C * H_C), THR, 0, stream>>>(hstate, 6 * B_C * H_C);

  // Prenet: relu(di@W1^T+b1) -> relu(@W2^T+b2), 64000 rows fully parallel.
  gemm_relu_f16<<<dim3(P_C / 16, (TDEC_C * B_C) / 128), THR, 0, stream>>>(
      dip, MK_C, MK_C, w_pw1, pre_b1, x1, P_C);
  gemm_relu_f16<<<dim3(P_C / 16, (TDEC_C * B_C) / 128), THR, 0, stream>>>(
      x1, P_C, P_C, w_pw2, pre_b2, xbuf, P_C);

  const size_t HS = (size_t)B_C * H_C;
  h16* h1b = hstate;          // [2][64][H]
  h16* h2b = hstate + 2 * HS;
  h16* h3b = hstate + 4 * HS;

  // -- Sequential decode: 5 dependent WMMA kernels per step --
  for (int t = 0; t < TDEC_C; ++t) {
    int p = t & 1, q = 1 - p;
    const h16* xt = xbuf + (size_t)t * B_C * P_C;
    const h16* mt = membuf + (size_t)t * B_C * E_C;

    // GRU1: x=[xt|mt] (K=768), h=h1 (K=1024)
    gru_step<<<H_C / 32, THR, 0, stream>>>(
        xt, P_C, P_C, mt, E_C, E_C, h1b + p * HS,
        w_g1ih, P_C + E_C, w_g1hh, g1_bih, g1_bhh,
        h1b + q * HS, nullptr, nullptr);
    // d2 = [h1|mt] @ att_w^T + att_b
    gemm2seg<<<H_C / 32, THR, 0, stream>>>(
        h1b + q * HS, H_C, H_C, mt, E_C, E_C,
        w_att, H_C + E_C, att_b, d2, nullptr, H_C);
    // GRU2 (+ r2 = h2 + d2)
    gru_step<<<H_C / 32, THR, 0, stream>>>(
        d2, H_C, H_C, nullptr, 0, 0, h2b + p * HS,
        w_g2ih, H_C, w_g2hh, g2_bih, g2_bhh,
        h2b + q * HS, d2, r2);
    // GRU3 (+ r3 = h3 + r2)
    gru_step<<<H_C / 32, THR, 0, stream>>>(
        r2, H_C, H_C, nullptr, 0, 0, h3b + p * HS,
        w_g3ih, H_C, w_g3hh, g3_bih, g3_bhh,
        h3b + q * HS, r2, r3);
    // out[:, t, :] = [r3|mt] @ proj_w^T + proj_b   (f32, straight to d_out)
    gemm2seg<<<(M_C * NFPS_C) / 32, THR, 0, stream>>>(
        r3, H_C, H_C, mt, E_C, E_C,
        w_proj, H_C + E_C, proj_b,
        nullptr, out + (size_t)t * M_C * NFPS_C, TDEC_C * M_C * NFPS_C);
  }
}
